// HeteroGNN_81716047774001
// MI455X (gfx1250) — compile-verified, matrix operand order checked by reading, not run on hardware
//
#include <hip/hip_runtime.h>

typedef __bf16 bf16;
typedef __attribute__((ext_vector_type(8)))  __bf16 v8bf;
typedef __attribute__((ext_vector_type(16))) __bf16 v16bf;
typedef __attribute__((ext_vector_type(8)))  float  v8f;
typedef __attribute__((ext_vector_type(4)))  float  v4f;

static constexpr int kD   = 256;
static constexpr int kNG  = 100000;
static constexpr int kNC  = 2000;
static constexpr int kE   = 1000000;
static constexpr int kL   = 2;
static constexpr int kOUT = 64;
static constexpr int LDK  = kD + 8;   // padded LDS K-stride (bf16) -> 528B rows, conflict-free b128

__device__ __forceinline__ bf16 f2bf(float f) { return (bf16)f; }   // native v_cvt
__device__ __forceinline__ float bf2f(bf16 b) {
    unsigned short s = __builtin_bit_cast(unsigned short, b);
    unsigned u = ((unsigned)s) << 16;
    return __builtin_bit_cast(float, u);
}

__device__ __forceinline__ v8f wmma_bf16(v16bf a, v16bf b, v8f c) {
    // v_wmma_f32_16x16x32_bf16: D = A(16x32) x B(32x16) + C(16x16 f32)
    return __builtin_amdgcn_wmma_f32_16x16x32_bf16(false, a, false, b, (short)0, c, false, false);
}

// 16-bit A 16x32 layout (ISA 7.12.2): lane<16 -> row=lane, K = base..base+7 and
// base+16..base+23, base = (lane>=16)*8. B (= weight rows for out=A*W^T) mirrors it.
__device__ __forceinline__ v16bf frag_from_bf16(const bf16* p) {
    v8bf lo = *(const v8bf*)(p);
    v8bf hi = *(const v8bf*)(p + 16);
    v16bf r;
#pragma unroll
    for (int i = 0; i < 8; i++) { r[i] = lo[i]; r[8 + i] = hi[i]; }
    return r;
}

// async global -> LDS staging of a 64-row x 256-col bf16 weight tile
// (GLOBAL_LOAD_ASYNC_TO_LDS_B128: per-lane LDS dest offset in VDST, tracked by ASYNCcnt)
__device__ __forceinline__ void fill_lds_async(bf16* dst, const bf16* W, int ncol0) {
#pragma unroll
    for (int c = 0; c < 8; c++) {
        int chunk = (int)threadIdx.x + c * 256;      // 2048 16B chunks
        int row = chunk >> 5, col8 = chunk & 31;
        unsigned ldsoff = (unsigned)(uintptr_t)(dst + row * LDK + col8 * 8);
        unsigned long long ga =
            (unsigned long long)(uintptr_t)(W + (size_t)(ncol0 + row) * kD + col8 * 8);
        asm volatile("global_load_async_to_lds_b128 %0, %1, off"
                     :: "v"(ldsoff), "v"(ga) : "memory");
    }
}
__device__ __forceinline__ void wait_async_all() {
    asm volatile("s_wait_asynccnt 0x0" ::: "memory");
}

// --- small utility kernels --------------------------------------------------
__global__ void k_cvt_bf16x8(const float* __restrict__ in, bf16* __restrict__ out, int n8) {
    int i = blockIdx.x * blockDim.x + threadIdx.x;
    if (i < n8) {
        v4f a = *((const v4f*)in + 2 * (size_t)i);
        v4f b = *((const v4f*)in + 2 * (size_t)i + 1);
        v8bf r;
#pragma unroll
        for (int j = 0; j < 4; j++) { r[j] = f2bf(a[j]); r[4 + j] = f2bf(b[j]); }
        *((v8bf*)out + i) = r;
    }
}

__global__ void k_comb_bf16x8(const float* __restrict__ a, const float* __restrict__ b,
                              bf16* __restrict__ out, int n8) {
    int i = blockIdx.x * blockDim.x + threadIdx.x;
    if (i < n8) {
        v4f a0 = *((const v4f*)a + 2 * (size_t)i);
        v4f a1 = *((const v4f*)a + 2 * (size_t)i + 1);
        v8bf r;
        if (b) {
            v4f b0 = *((const v4f*)b + 2 * (size_t)i);
            v4f b1 = *((const v4f*)b + 2 * (size_t)i + 1);
#pragma unroll
            for (int j = 0; j < 4; j++) { r[j] = f2bf(a0[j] + b0[j]); r[4 + j] = f2bf(a1[j] + b1[j]); }
        } else {
#pragma unroll
            for (int j = 0; j < 4; j++) { r[j] = f2bf(a0[j]); r[4 + j] = f2bf(a1[j]); }
        }
        *((v8bf*)out + i) = r;
    }
}

__global__ void k_addf32(const float* __restrict__ a, const float* __restrict__ b,
                         float* __restrict__ out, int n) {
    int i = blockIdx.x * blockDim.x + threadIdx.x;
    if (i < n) out[i] = a[i] + b[i];
}

__global__ void k_zero4(float4* __restrict__ p, int n4) {
    int i = blockIdx.x * blockDim.x + threadIdx.x;
    if (i < n4) p[i] = make_float4(0.f, 0.f, 0.f, 0.f);
}

// --- edge aggregation: agg[dst] += bf2f(x[src]) * w  (fp32 accumulate) ------
// one edge per wave32; each lane handles 8 contiguous elements (b128 gather)
__global__ void __launch_bounds__(256) k_edge_agg(const bf16* __restrict__ x,
                                                  const int* __restrict__ src,
                                                  const int* __restrict__ dst,
                                                  const float* __restrict__ w,
                                                  float* __restrict__ agg, int E) {
    int e = blockIdx.x * 8 + (threadIdx.x >> 5);
    if (e >= E) return;
    int lane = threadIdx.x & 31;
    int s = src[e], d = dst[e];
    float wt = w[e];
    v8bf v = *((const v8bf*)(x + (size_t)s * kD) + lane);
    float* out = agg + (size_t)d * kD + lane * 8;
#pragma unroll
    for (int i = 0; i < 8; i++) atomicAdd(out + i, bf2f(v[i]) * wt);
}

// --- fused WMMA GEMM --------------------------------------------------------
// out = relu( aggB @ Wa^T + x @ Ws^T + diag(wrow) * (x @ Wb^T) + bias ), bf16 out
// block: 256 thr (8 waves); tile 128(M) x 64(N); all 3 weight tiles async-staged
// in LDS (3 x ~33KB of the 320KB WGP LDS); single K loop, 12 WMMA per K-step.
__global__ void __launch_bounds__(256) k_gemm_fused(const bf16* __restrict__ aggB,
                                                    const bf16* __restrict__ x,
                                                    const float* __restrict__ wrow,
                                                    const bf16* __restrict__ Wa,
                                                    const bf16* __restrict__ Ws,
                                                    const bf16* __restrict__ Wb,
                                                    const float* __restrict__ bias,
                                                    bf16* __restrict__ out, int M) {
    __shared__ bf16 lds[3][64 * LDK];
    const int lane = threadIdx.x & 31;
    const int wave = threadIdx.x >> 5;
    const int lr   = lane & 15;
    const int half = lane >> 4;
    const int mrow = blockIdx.x * 128 + wave * 16;
    const int ncol0 = blockIdx.y * 64;
    int ar = mrow + lr;
    if (ar > M - 1) ar = M - 1;            // clamp loads; EXEC stays full for WMMA
    const int kb = half * 8;

    fill_lds_async(&lds[0][0], Wa, ncol0);
    fill_lds_async(&lds[1][0], Ws, ncol0);
    fill_lds_async(&lds[2][0], Wb, ncol0);
    wait_async_all();
    __syncthreads();

    v8f accm[4] = {};                      // aggB@Wa + x@Ws
    v8f accb[4] = {};                      // x@Wb (row-scaled in epilogue)
    const bf16* aRow = aggB + (size_t)ar * kD + kb;
    const bf16* xRow = x    + (size_t)ar * kD + kb;
#pragma unroll
    for (int k0 = 0; k0 < kD; k0 += 32) {
        v16bf aA = frag_from_bf16(aRow + k0);
        v16bf aX = frag_from_bf16(xRow + k0);
#pragma unroll
        for (int n = 0; n < 4; n++) {
            const int boff = (n * 16 + lr) * LDK + k0 + kb;
            accm[n] = wmma_bf16(aA, frag_from_bf16(&lds[0][0] + boff), accm[n]);
            accm[n] = wmma_bf16(aX, frag_from_bf16(&lds[1][0] + boff), accm[n]);
            accb[n] = wmma_bf16(aX, frag_from_bf16(&lds[2][0] + boff), accb[n]);
        }
    }

    // epilogue: C/D layout VGPR i, lanes0-15 -> M=i, lanes16-31 -> M=i+8
#pragma unroll
    for (int n = 0; n < 4; n++) {
        int ncol = ncol0 + n * 16 + lr;
        float bn = bias[ncol];
#pragma unroll
        for (int i = 0; i < 8; i++) {
            int m = mrow + i + half * 8;
            if (m < M) {
                float v = accm[n][i] + wrow[m] * accb[n][i] + bn;
                v = v > 0.f ? v : 0.f;
                out[(size_t)m * kD + ncol] = f2bf(v);
            }
        }
    }
}

// --- final projection: out(f32) = x @ Wout^T + bout  (M x 64) ---------------
__global__ void __launch_bounds__(256) k_out_gemm(const bf16* __restrict__ x,
                                                  const bf16* __restrict__ W,
                                                  const float* __restrict__ bout,
                                                  float* __restrict__ out, int M) {
    __shared__ bf16 lds[64 * LDK];
    const int lane = threadIdx.x & 31;
    const int wave = threadIdx.x >> 5;
    const int lr   = lane & 15;
    const int half = lane >> 4;
    const int mrow = blockIdx.x * 128 + wave * 16;
    int ar = mrow + lr;
    if (ar > M - 1) ar = M - 1;
    const int kb = half * 8;

    fill_lds_async(lds, W, 0);
    wait_async_all();
    __syncthreads();

    v8f acc[4] = {};
    const bf16* xRow = x + (size_t)ar * kD + kb;
#pragma unroll
    for (int k0 = 0; k0 < kD; k0 += 32) {
        v16bf a = frag_from_bf16(xRow + k0);
#pragma unroll
        for (int n = 0; n < 4; n++) {
            v16bf b = frag_from_bf16(lds + (n * 16 + lr) * LDK + k0 + kb);
            acc[n] = wmma_bf16(a, b, acc[n]);
        }
    }
#pragma unroll
    for (int n = 0; n < 4; n++) {
        int ncol = n * 16 + lr;
        float bn = bout[ncol];
#pragma unroll
        for (int i = 0; i < 8; i++) {
            int m = mrow + i + half * 8;
            if (m < M) out[(size_t)m * kOUT + ncol] = acc[n][i] + bn;
        }
    }
}

extern "C" void kernel_launch(void* const* d_in, const int* in_sizes, int n_in,
                              void* d_out, int out_size, void* d_ws, size_t ws_size,
                              hipStream_t stream) {
    const float* x_gene = (const float*)d_in[0];
    const float* x_cell = (const float*)d_in[1];
    const int*   src_g2c = (const int*)d_in[2];
    const int*   dst_g2c = (const int*)d_in[3];
    const int*   src_c2g = (const int*)d_in[4];
    const int*   dst_c2g = (const int*)d_in[5];
    const float* w_g2c = (const float*)d_in[8];
    const float* w_c2g = (const float*)d_in[9];
    const float* w_gg  = (const float*)d_in[10];
    const float* w_cc  = (const float*)d_in[11];
    const float* Wrel  = (const float*)d_in[12];
    const float* brel  = (const float*)d_in[13];
    const float* Wroot = (const float*)d_in[14];
    const float* Wout  = (const float*)d_in[15];
    const float* bout  = (const float*)d_in[16];

    char* ws = (char*)d_ws;
    size_t off = 0;
    auto alloc = [&](size_t bytes) -> char* {
        char* p = ws + off;
        off += (bytes + 255) & ~(size_t)255;
        return p;
    };
    bf16*  xgA   = (bf16*)alloc((size_t)kNG * kD * 2);
    bf16*  xgB   = (bf16*)alloc((size_t)kNG * kD * 2);
    bf16*  xcA   = (bf16*)alloc((size_t)kNC * kD * 2);
    bf16*  xcB   = (bf16*)alloc((size_t)kNC * kD * 2);
    float* aggG  = (float*)alloc((size_t)kNG * kD * 4);
    float* aggC  = (float*)alloc((size_t)kNC * kD * 4);
    bf16*  aggGB = (bf16*)alloc((size_t)kNG * kD * 2);
    bf16*  aggCB = (bf16*)alloc((size_t)kNC * kD * 2);
    bf16*  Wb[kL][6];    // per layer: WrelC_a, WsumC, WrelC_b, WrelG_a, WsumG, WrelG_b
    float* biasC[kL];
    float* biasG[kL];
    for (int l = 0; l < kL; l++) {
        for (int i = 0; i < 6; i++) Wb[l][i] = (bf16*)alloc((size_t)kD * kD * 2);
        biasC[l] = (float*)alloc(kD * 4);
        biasG[l] = (float*)alloc(kD * 4);
    }
    bf16* WoB = (bf16*)alloc((size_t)kOUT * kD * 2);

    const int T = 256;
    const int nW = kD * kD;

    // activations -> bf16
    {
        int n8 = kNG * kD / 8;
        k_cvt_bf16x8<<<(n8 + T - 1) / T, T, 0, stream>>>(x_gene, xgA, n8);
        n8 = kNC * kD / 8;
        k_cvt_bf16x8<<<(n8 + T - 1) / T, T, 0, stream>>>(x_cell, xcA, n8);
    }
    // weight prep: fold Wroot pairs + biases; convert to bf16
    for (int l = 0; l < kL; l++) {
        const float* Wr = Wrel  + (size_t)l * 4 * nW;
        const float* Wo = Wroot + (size_t)l * 4 * nW;
        int n8 = nW / 8, gw = (n8 + T - 1) / T;
        // cell target: rel0, root0+root3, rel3
        k_comb_bf16x8<<<gw, T, 0, stream>>>(Wr + 0 * (size_t)nW, nullptr,             Wb[l][0], n8);
        k_comb_bf16x8<<<gw, T, 0, stream>>>(Wo + 0 * (size_t)nW, Wo + 3 * (size_t)nW, Wb[l][1], n8);
        k_comb_bf16x8<<<gw, T, 0, stream>>>(Wr + 3 * (size_t)nW, nullptr,             Wb[l][2], n8);
        // gene target: rel1, root1+root2, rel2
        k_comb_bf16x8<<<gw, T, 0, stream>>>(Wr + 1 * (size_t)nW, nullptr,             Wb[l][3], n8);
        k_comb_bf16x8<<<gw, T, 0, stream>>>(Wo + 1 * (size_t)nW, Wo + 2 * (size_t)nW, Wb[l][4], n8);
        k_comb_bf16x8<<<gw, T, 0, stream>>>(Wr + 2 * (size_t)nW, nullptr,             Wb[l][5], n8);
        const float* br = brel + (size_t)l * 4 * kD;
        k_addf32<<<1, T, 0, stream>>>(br + 0 * kD, br + 3 * kD, biasC[l], kD);
        k_addf32<<<1, T, 0, stream>>>(br + 1 * kD, br + 2 * kD, biasG[l], kD);
    }
    {
        int n8 = kOUT * kD / 8;
        k_comb_bf16x8<<<(n8 + T - 1) / T, T, 0, stream>>>(Wout, nullptr, WoB, n8);
    }

    bf16 *xg = xgA, *xgN = xgB, *xc = xcA, *xcN = xcB;
    for (int l = 0; l < kL; l++) {
        int n4c = kNC * kD / 4, n4g = kNG * kD / 4;
        k_zero4<<<(n4c + T - 1) / T, T, 0, stream>>>((float4*)aggC, n4c);
        k_zero4<<<(n4g + T - 1) / T, T, 0, stream>>>((float4*)aggG, n4g);

        k_edge_agg<<<(kE + 7) / 8, T, 0, stream>>>(xg, src_g2c, dst_g2c, w_g2c, aggC, kE);
        k_edge_agg<<<(kE + 7) / 8, T, 0, stream>>>(xc, src_c2g, dst_c2g, w_c2g, aggG, kE);

        int n8c = kNC * kD / 8, n8g = kNG * kD / 8;
        k_cvt_bf16x8<<<(n8c + T - 1) / T, T, 0, stream>>>(aggC, aggCB, n8c);
        k_cvt_bf16x8<<<(n8g + T - 1) / T, T, 0, stream>>>(aggG, aggGB, n8g);

        dim3 gc((kNC + 127) / 128, kD / 64);
        k_gemm_fused<<<gc, T, 0, stream>>>(aggCB, xc, w_cc, Wb[l][0], Wb[l][1], Wb[l][2],
                                           biasC[l], xcN, kNC);
        dim3 gg((kNG + 127) / 128, kD / 64);
        k_gemm_fused<<<gg, T, 0, stream>>>(aggGB, xg, w_gg, Wb[l][3], Wb[l][4], Wb[l][5],
                                           biasG[l], xgN, kNG);
        bf16* t;
        t = xg; xg = xgN; xgN = t;
        t = xc; xc = xcN; xcN = t;
    }
    k_out_gemm<<<(kNC + 127) / 128, T, 0, stream>>>(xc, WoB, bout, (float*)d_out, kNC);
}
